// ManifoldConstrainedHyperConnection_57140244906290
// MI455X (gfx1250) — compile-verified, matrix operand order checked by reading, not run on hardware
//
#include <hip/hip_runtime.h>
#include <hip/hip_bf16.h>

// ---------------------------------------------------------------------------
// Problem constants (from reference)
// ---------------------------------------------------------------------------
#define L_LAYERS 12
#define B_BATCH  8
#define S_SEQ    2048
#define H_DIM    1024
#define MD_DIM   32
#define D_DIM    256            // H/4
#define GAMMA_C  0.9f
#define BETA_C   0.5f
#define LN_EPS_C 1e-5f

// GEMM tiling
#define TILE_M 128
#define TILE_N 128
#define TILE_K 32
#define KSTEPS      (H_DIM / TILE_K)          // 32
#define TOTAL_STEPS (L_LAYERS * KSTEPS)       // 384
#define LDA_P  40               // padded bf16 row stride, 80B rows (16B aligned, bank-clean)
#define LDB_P  40

typedef __attribute__((ext_vector_type(16))) __bf16   v16bf;
typedef __attribute__((ext_vector_type(2)))  __bf16   v2bf;
typedef __attribute__((ext_vector_type(8)))  float    v8f;
typedef __attribute__((ext_vector_type(8)))  unsigned v8u;

// pointer types matching the async-LDS builtin prototype (probe-confirmed):
//   param0: global (AS1) v4i*   param1: local (AS3) v4i*
typedef int v4i_vs __attribute__((vector_size(16)));
typedef __attribute__((address_space(1))) v4i_vs* as1_v4i_p;
typedef __attribute__((address_space(3))) v4i_vs* as3_v4i_p;

__device__ __forceinline__ unsigned short f32_to_bf16_rne(float f) {
    unsigned u = __builtin_bit_cast(unsigned, f);
    unsigned r = 0x7FFFu + ((u >> 16) & 1u);
    return (unsigned short)((u + r) >> 16);
}

// pack two f32 into packed bf16x2 (single V_CVT_PK_BF16_F32 when available)
__device__ __forceinline__ unsigned pack2_bf16(float x, float y) {
#if __has_builtin(__builtin_amdgcn_cvt_pk_bf16_f32)
    v2bf r = __builtin_amdgcn_cvt_pk_bf16_f32(x, y);
    return __builtin_bit_cast(unsigned, r);
#else
    return (unsigned)f32_to_bf16_rne(x) | ((unsigned)f32_to_bf16_rne(y) << 16);
#endif
}

// ---- CDNA5 async global->LDS copy (ASYNCcnt-tracked) ----------------------
__device__ __forceinline__ void async_copy_b128(const void* g, void* l) {
#if __has_builtin(__builtin_amdgcn_global_load_async_to_lds_b128)
    __builtin_amdgcn_global_load_async_to_lds_b128(
        (as1_v4i_p)g, (as3_v4i_p)l, 0, 0);
#else
    unsigned loff = (unsigned)(unsigned long long)l;
    unsigned long long ga = (unsigned long long)g;
    asm volatile("global_load_async_to_lds_b128 %0, %1, off"
                 :: "v"(loff), "v"(ga) : "memory");
#endif
}

__device__ __forceinline__ void wait_async0() {
#if __has_builtin(__builtin_amdgcn_s_wait_asynccnt)
    __builtin_amdgcn_s_wait_asynccnt(0);
#else
    asm volatile("s_wait_asynccnt 0" ::: "memory");
#endif
}

// ---------------------------------------------------------------------------
// 1) connection vector m = softmax(softmax((1-B)*W[cur] + B*(U@V)[cur]))
// ---------------------------------------------------------------------------
__global__ void m_kernel(const float* __restrict__ CW, const float* __restrict__ U,
                         const float* __restrict__ V, const int* __restrict__ curp,
                         float* __restrict__ mout) {
    __shared__ float sm[L_LAYERS];
    const int t = threadIdx.x;
    const int cur = *curp;
    if (t < L_LAYERS) {
        float acc = 0.f;
        #pragma unroll
        for (int k = 0; k < MD_DIM; ++k)
            acc += U[cur * MD_DIM + k] * V[k * L_LAYERS + t];
        sm[t] = (1.0f - BETA_C) * CW[cur * L_LAYERS + t] + BETA_C * acc;
    }
    __syncthreads();
    if (t == 0) {
        #pragma unroll
        for (int pass = 0; pass < 2; ++pass) {
            float mx = -3.4e38f;
            for (int l = 0; l < L_LAYERS; ++l) mx = fmaxf(mx, sm[l]);
            float se = 0.f;
            for (int l = 0; l < L_LAYERS; ++l) { float e = __expf(sm[l] - mx); sm[l] = e; se += e; }
            float inv = 1.0f / se;
            for (int l = 0; l < L_LAYERS; ++l) sm[l] *= inv;
        }
        for (int l = 0; l < L_LAYERS; ++l) mout[l] = sm[l];
    }
}

// ---------------------------------------------------------------------------
// 2) pooled[l,b,h] = mean_s X[l,b,s,h]
// ---------------------------------------------------------------------------
__global__ __launch_bounds__(256)
void pool_kernel(const float* __restrict__ X, float* __restrict__ pooled) {
    const int gid = blockIdx.x * 256 + threadIdx.x;
    const float* p = X + (size_t)(gid / H_DIM) * S_SEQ * H_DIM + (gid % H_DIM);
    float s = 0.f;
    for (int si = 0; si < S_SEQ; ++si) s += p[(size_t)si * H_DIM];
    pooled[gid] = s * (1.0f / (float)S_SEQ);
}

// ---------------------------------------------------------------------------
// 2b) WpT[l][n][k] = bf16(proj_W[l][k][n])  -- K-contiguous transposed copy
// ---------------------------------------------------------------------------
__global__ __launch_bounds__(256)
void wpT_kernel(const float* __restrict__ Wp, unsigned short* __restrict__ WpT) {
    __shared__ float tile[32][33];
    const int l  = blockIdx.z;
    const int h0 = blockIdx.y * 32;
    const int n0 = blockIdx.x * 32;
    const int tr = threadIdx.x >> 5, tc = threadIdx.x & 31;
    const float* src = Wp + (size_t)l * H_DIM * H_DIM;
    #pragma unroll
    for (int i = 0; i < 4; ++i) {
        const int r = tr + i * 8;
        tile[r][tc] = src[(size_t)(h0 + r) * H_DIM + n0 + tc];
    }
    __syncthreads();
    unsigned short* dst = WpT + (size_t)l * H_DIM * H_DIM;
    #pragma unroll
    for (int i = 0; i < 4; ++i) {
        const int r = tr + i * 8;
        dst[(size_t)(n0 + r) * H_DIM + h0 + tc] = f32_to_bf16_rne(tile[tc][r]);
    }
}

// ---------------------------------------------------------------------------
// 3) gate MLP per (l,b): Linear -> LN -> GELU(exact) -> Linear -> sigmoid
// ---------------------------------------------------------------------------
__global__ __launch_bounds__(256)
void gate_kernel(const float* __restrict__ pooled, const float* __restrict__ W1,
                 const float* __restrict__ b1, const float* __restrict__ lnw,
                 const float* __restrict__ lnb, const float* __restrict__ W2,
                 const float* __restrict__ b2, const float* __restrict__ mvec,
                 float* __restrict__ wout) {
    __shared__ float sp[H_DIM];
    __shared__ float red[D_DIM];
    const int lb = blockIdx.x;
    const int l  = lb / B_BATCH;
    const int t  = threadIdx.x;

    for (int i = t; i < H_DIM; i += 256) sp[i] = pooled[(size_t)lb * H_DIM + i];
    __syncthreads();

    const float* W1l = W1 + (size_t)l * H_DIM * D_DIM;
    float hv = b1[l * D_DIM + t];
    for (int hh = 0; hh < H_DIM; ++hh) hv += sp[hh] * W1l[(size_t)hh * D_DIM + t];

    red[t] = hv; __syncthreads();
    for (int off = 128; off > 0; off >>= 1) { if (t < off) red[t] += red[t + off]; __syncthreads(); }
    const float mu = red[0] * (1.0f / (float)D_DIM);
    __syncthreads();
    const float dv = hv - mu;
    red[t] = dv * dv; __syncthreads();
    for (int off = 128; off > 0; off >>= 1) { if (t < off) red[t] += red[t + off]; __syncthreads(); }
    const float var = red[0] * (1.0f / (float)D_DIM);
    __syncthreads();

    const float hn = dv * rsqrtf(var + LN_EPS_C) * lnw[l * D_DIM + t] + lnb[l * D_DIM + t];
    const float hg = 0.5f * hn * (1.0f + erff(hn * 0.70710678118654752f));

    red[t] = hg * W2[l * D_DIM + t]; __syncthreads();
    for (int off = 128; off > 0; off >>= 1) { if (t < off) red[t] += red[t + off]; __syncthreads(); }
    if (t == 0) {
        const float g = 1.0f / (1.0f + __expf(-(red[0] + b2[l])));
        wout[lb] = mvec[l] * g;
    }
}

// ---------------------------------------------------------------------------
// 4) bsum[b,k] = sum_l w[l,b] * proj_b[l,k]
// ---------------------------------------------------------------------------
__global__ void bsum_kernel(const float* __restrict__ wlb, const float* __restrict__ pb,
                            float* __restrict__ bsum) {
    const int b = blockIdx.y;
    const int k = blockIdx.x * 256 + threadIdx.x;
    float s = 0.f;
    #pragma unroll
    for (int l = 0; l < L_LAYERS; ++l) s += wlb[l * B_BATCH + b] * pb[l * H_DIM + k];
    bsum[b * H_DIM + k] = s;
}

// ---------------------------------------------------------------------------
// 5) Fused weighted GEMM, bf16 WMMA, double-buffered LDS.
//    A: f32 X scaled by w[l,b], packed to bf16 (V_CVT_PK_BF16_F32) in-kernel.
//    B: bf16 WpT tiles copied with GLOBAL_LOAD_ASYNC_TO_LDS_B128 (ASYNCcnt).
// ---------------------------------------------------------------------------
__global__ __launch_bounds__(256)
void hyper_gemm(const float* __restrict__ X, const unsigned short* __restrict__ WpT,
                const float* __restrict__ wlb, const float* __restrict__ bsum,
                const int* __restrict__ curp, float* __restrict__ out) {
    __shared__ unsigned short lA [2][TILE_M * LDA_P];
    __shared__ unsigned short lBT[2][TILE_N * LDB_P];
    __shared__ float sw[L_LAYERS];

    const int b  = blockIdx.z;
    const int s0 = blockIdx.y * TILE_M;
    const int n0 = blockIdx.x * TILE_N;
    const int t    = threadIdx.x;
    const int lane = t & 31;
    const int wid  = t >> 5;
    const int wm   = wid & 3;        // 4 x 32 rows
    const int wn   = wid >> 2;       // 2 x 64 cols
    const int mlane = lane & 15;
    const int khalf = lane >> 4;

    if (t < L_LAYERS) sw[t] = wlb[t * B_BATCH + b];
    __syncthreads();

    v8f acc[2][4] = {};

    auto stage = [&](int step, int buf) {
        const int l  = step >> 5;                 // KSTEPS == 32
        const int k0 = (step & 31) * TILE_K;
        const float wa = sw[l];
        const float* Xl = X + ((size_t)l * B_BATCH + b) * (size_t)S_SEQ * H_DIM;
        // --- A tile (128 x 32): f32 load -> scale -> packed bf16 -> LDS ---
        #pragma unroll
        for (int it = 0; it < 4; ++it) {
            const int idx = (t + it * 256) * 4;
            const int row = idx >> 5;
            const int kk  = idx & 31;
            const float4 v = *(const float4*)(Xl + (size_t)(s0 + row) * H_DIM + k0 + kk);
            const unsigned p0 = pack2_bf16(v.x * wa, v.y * wa);
            const unsigned p1 = pack2_bf16(v.z * wa, v.w * wa);
            *(unsigned*)&lA[buf][row * LDA_P + kk]     = p0;
            *(unsigned*)&lA[buf][row * LDA_P + kk + 2] = p1;
        }
        // --- B tile (128n x 32k bf16): async 16B chunks straight into LDS ---
        #pragma unroll
        for (int c = 0; c < 2; ++c) {
            const int chunk = t * 2 + c;          // 512 chunks of 16B
            const int n     = chunk >> 2;
            const int koff  = (chunk & 3) * 8;
            const unsigned short* gp =
                WpT + ((size_t)(l * H_DIM + n0 + n) * H_DIM + k0 + koff);
            async_copy_b128(gp, &lBT[buf][n * LDB_P + koff]);
        }
    };

    stage(0, 0);

    for (int step = 0; step < TOTAL_STEPS; ++step) {
        const int buf = step & 1;
        wait_async0();          // our async B writes for this buffer are done
        __syncthreads();        // everyone's A stores done; prev buffer free
        if (step + 1 < TOTAL_STEPS) stage(step + 1, buf ^ 1);

        // --- gather fragments (16-bit A/B VGPR layout) ---
        // VGPR v holds K pair base: v<4 -> 2v else 16+2(v-4); +8 for lanes 16..31
        v16bf afr[2], bfr[4];
        #pragma unroll
        for (int fm = 0; fm < 2; ++fm) {
            const int row = wm * 32 + fm * 16 + mlane;
            v8u tmp;
            #pragma unroll
            for (int v = 0; v < 8; ++v) {
                const int kb = ((v < 4) ? (2 * v) : (16 + 2 * (v - 4))) + khalf * 8;
                tmp[v] = *(const unsigned*)&lA[buf][row * LDA_P + kb];
            }
            afr[fm] = __builtin_bit_cast(v16bf, tmp);
        }
        #pragma unroll
        for (int fn = 0; fn < 4; ++fn) {
            const int col = wn * 64 + fn * 16 + mlane;
            v8u tmp;
            #pragma unroll
            for (int v = 0; v < 8; ++v) {
                const int kb = ((v < 4) ? (2 * v) : (16 + 2 * (v - 4))) + khalf * 8;
                tmp[v] = *(const unsigned*)&lBT[buf][col * LDB_P + kb];
            }
            bfr[fn] = __builtin_bit_cast(v16bf, tmp);
        }

        #pragma unroll
        for (int fm = 0; fm < 2; ++fm)
            #pragma unroll
            for (int fn = 0; fn < 4; ++fn)
                acc[fm][fn] = __builtin_amdgcn_wmma_f32_16x16x32_bf16(
                    false, afr[fm], false, bfr[fn],
                    (short)0, acc[fm][fn], false, false);
    }

    // --- epilogue: GAMMA * (acc + bias) + residual X[cur] ---
    const int cur = *curp;
    const float* Xc = X + ((size_t)cur * B_BATCH + b) * (size_t)S_SEQ * H_DIM;
    const int mhalf = khalf * 8;
    #pragma unroll
    for (int fm = 0; fm < 2; ++fm) {
        #pragma unroll
        for (int fn = 0; fn < 4; ++fn) {
            const int col = n0 + wn * 64 + fn * 16 + mlane;
            const float bs = bsum[b * H_DIM + col];
            #pragma unroll
            for (int r = 0; r < 8; ++r) {
                const int srow = s0 + wm * 32 + fm * 16 + mhalf + r;
                const float v = acc[fm][fn][r] + bs;
                out[((size_t)b * S_SEQ + srow) * H_DIM + col] =
                    GAMMA_C * v + Xc[(size_t)srow * H_DIM + col];
            }
        }
    }
}

// ---------------------------------------------------------------------------
// launcher
// ---------------------------------------------------------------------------
extern "C" void kernel_launch(void* const* d_in, const int* in_sizes, int n_in,
                              void* d_out, int out_size, void* d_ws, size_t ws_size,
                              hipStream_t stream) {
    const float* X   = (const float*)d_in[0];
    const float* CW  = (const float*)d_in[1];
    const float* U   = (const float*)d_in[2];
    const float* V   = (const float*)d_in[3];
    const float* W1  = (const float*)d_in[4];
    const float* b1  = (const float*)d_in[5];
    const float* lnw = (const float*)d_in[6];
    const float* lnb = (const float*)d_in[7];
    const float* W2  = (const float*)d_in[8];
    const float* b2  = (const float*)d_in[9];
    const float* Wp  = (const float*)d_in[10];
    const float* pb  = (const float*)d_in[11];
    const int*   cur = (const int*)d_in[12];
    float* out = (float*)d_out;

    float* ws      = (float*)d_ws;
    float* m_ws    = ws;                                        // 12
    float* w_ws    = ws + 16;                                   // 96
    float* pool_ws = ws + 128;                                  // L*B*H
    float* bsum_ws = ws + 128 + L_LAYERS * B_BATCH * H_DIM;     // B*H
    unsigned short* wpT_ws =
        (unsigned short*)(ws + 128 + L_LAYERS * B_BATCH * H_DIM + B_BATCH * H_DIM + 64);

    m_kernel<<<1, 32, 0, stream>>>(CW, U, V, cur, m_ws);
    pool_kernel<<<(L_LAYERS * B_BATCH * H_DIM) / 256, 256, 0, stream>>>(X, pool_ws);
    wpT_kernel<<<dim3(H_DIM / 32, H_DIM / 32, L_LAYERS), 256, 0, stream>>>(Wp, wpT_ws);
    gate_kernel<<<L_LAYERS * B_BATCH, 256, 0, stream>>>(pool_ws, W1, b1, lnw, lnb,
                                                        W2, b2, m_ws, w_ws);
    bsum_kernel<<<dim3(H_DIM / 256, B_BATCH), 256, 0, stream>>>(w_ws, pb, bsum_ws);
    hyper_gemm<<<dim3(H_DIM / TILE_N, S_SEQ / TILE_M, B_BATCH), 256, 0, stream>>>(
        X, wpT_ws, w_ws, bsum_ws, cur, out);
}
